// RNNmodel_15960098472576
// MI455X (gfx1250) — matrix operand order repaired and checked by reading, not compiled, not run on hardware
//
#include <hip/hip_runtime.h>
#include <cstdint>

// ---------------------------------------------------------------------------
// Shapes
// ---------------------------------------------------------------------------
#define Bb   32
#define Ss   256
#define DIN  512
#define DH   1024
#define KK   (DIN + DH)      // 1536
#define VV   32000
#define ROWS (Bb * Ss)       // 8192
#define NWG  16              // workgroups in persistent scan kernel

typedef __attribute__((ext_vector_type(16))) __bf16 v16bf;
typedef __attribute__((ext_vector_type(8)))  __bf16 v8bf;
typedef __attribute__((ext_vector_type(8)))  float  v8f;
typedef __attribute__((ext_vector_type(4)))  float  v4f;

// ---------------------------------------------------------------------------
// WMMA helpers (CDNA5, wave32): D(16x16,f32) = A(16x32,bf16) x B(32x16,bf16) + C
// ---------------------------------------------------------------------------
__device__ __forceinline__ v8f wmma_bf16(v16bf a, v16bf b, v8f c) {
  return __builtin_amdgcn_wmma_f32_16x16x32_bf16(
      /*neg_a=*/false, a, /*neg_b=*/false, b,
      /*c_mod=*/(short)0, c, /*reuse_a=*/false, /*reuse_b=*/false);
}

// A-operand: lane holds row M=L&15; elems 0..7  = K[kb + 8*lh .. +8),
//            elems 8..15 = K[kb + 16 + 8*lh .. +8)   (ISA 16-bit A 16x32 table)
__device__ __forceinline__ v16bf load_a(const __bf16* p, int lh) {
  union { v16bf v; v8bf h[2]; } u;
  u.h[0] = *(const v8bf*)(p + lh * 8);
  u.h[1] = *(const v8bf*)(p + 16 + lh * 8);
  return u.v;
}

// B-operand: lane holds col N=L&15; elems 0..7 = K[kb + 8*lh .. +8),
//            elems 8..15 = K[kb + 16 + 8*lh .. +8)   (same order as A)
__device__ __forceinline__ v16bf load_b(const __bf16* p, int lh) {
  union { v16bf v; v8bf h[2]; } u;
  u.h[0] = *(const v8bf*)(p + lh * 8);
  u.h[1] = *(const v8bf*)(p + 16 + lh * 8);
  return u.v;
}

// ---------------------------------------------------------------------------
// Init: h state from h0, zero LN stats + sync counters
// ---------------------------------------------------------------------------
__global__ __launch_bounds__(256) void init_kernel(const float* __restrict__ h0,
                                                   __bf16* __restrict__ hbuf,
                                                   float* __restrict__ stats,
                                                   unsigned* __restrict__ syncp) {
  int tid = blockIdx.x * 256 + threadIdx.x;
  if (tid < Bb * DH)     hbuf[tid]  = (__bf16)h0[tid & (DH - 1)];
  if (tid < Ss * Bb * 2) stats[tid] = 0.0f;
  if (tid < 64)          syncp[tid] = 0u;
}

// ---------------------------------------------------------------------------
// Embedding gather -> bf16   x[(b*S+s)*512 + k]
// ---------------------------------------------------------------------------
__global__ __launch_bounds__(256) void embed_kernel(const int* __restrict__ tokens,
                                                    const float* __restrict__ emb,
                                                    __bf16* __restrict__ xb) {
  size_t tid = (size_t)blockIdx.x * 256 + threadIdx.x;   // ROWS*DIN threads
  int row = (int)(tid >> 9);
  int k   = (int)(tid & (DIN - 1));
  int tok = tokens[row];
  xb[tid] = (__bf16)emb[(size_t)tok * DIN + k];
}

// ---------------------------------------------------------------------------
// Tiled transpose + f32->bf16 convert: src[R][C] -> dst[C][R]
// ---------------------------------------------------------------------------
__global__ __launch_bounds__(256) void tconv(const float* __restrict__ src,
                                             __bf16* __restrict__ dst,
                                             int R, int C) {
  __shared__ float tile[32][33];
  int c0 = blockIdx.x * 32, r0 = blockIdx.y * 32;
  for (int i = threadIdx.y; i < 32; i += 8)
    tile[i][threadIdx.x] = src[(size_t)(r0 + i) * C + c0 + threadIdx.x];
  __syncthreads();
  for (int i = threadIdx.y; i < 32; i += 8)
    dst[(size_t)(c0 + i) * R + r0 + threadIdx.x] = (__bf16)tile[threadIdx.x][i];
}

// ---------------------------------------------------------------------------
// Device-scope generation barrier for the persistent scan kernel
// ---------------------------------------------------------------------------
__device__ __forceinline__ void grid_sync(unsigned* cnt, unsigned* gen, unsigned& lgen) {
  __syncthreads();
  ++lgen;
  if (threadIdx.x == 0) {
    __threadfence();                                   // release prior writes
    unsigned prev = atomicAdd(cnt, 1u);
    if (prev == NWG - 1u) {
      __hip_atomic_store(cnt, 0u, __ATOMIC_RELAXED, __HIP_MEMORY_SCOPE_AGENT);
      __threadfence();
      atomicAdd(gen, 1u);
    } else {
      while (__hip_atomic_load(gen, __ATOMIC_ACQUIRE, __HIP_MEMORY_SCOPE_AGENT) < lgen)
        __builtin_amdgcn_s_sleep(4);
    }
    __threadfence();                                   // acquire
  }
  __syncthreads();
}

// ---------------------------------------------------------------------------
// Persistent recurrent scan: 16 WGs x 8 waves, 1 WMMA tile per wave per step.
// y = [x_t | h] @ W + b ; LayerNorm ; ReLU ; publish h (bf16) + hs row slice.
// K loop split into x-part (16 iters) and h-part (32 iters): no branches in
// the hot loop, so the compiler can clause the loads around the WMMAs.
// ---------------------------------------------------------------------------
__global__ __launch_bounds__(256) void rnn_scan(const __bf16* __restrict__ xb,
                                                const __bf16* __restrict__ wT,   // [DH][KK]
                                                const float*  __restrict__ bias,
                                                const float*  __restrict__ g_,
                                                const float*  __restrict__ bt_,
                                                __bf16* hbuf,                    // [Bb][DH]
                                                __bf16* __restrict__ hsb,        // [ROWS][DH]
                                                float* stats,                    // [Ss][Bb][2]
                                                unsigned* syncp) {
  const int lane = threadIdx.x & 31;
  const int wave = threadIdx.x >> 5;
  const int lh   = lane >> 4;          // half of wave
  const int tcol = lane & 15;
  const int mtile = wave & 1;                         // batches 0-15 / 16-31
  const int ntile = blockIdx.x * 4 + (wave >> 1);     // 0..63
  const int m0 = mtile * 16, n0 = ntile * 16;
  const int am = m0 + tcol;                           // A row (batch) loaded by this lane
  const int nn = n0 + tcol;                           // output column of this lane
  const __bf16* pb  = wT + (size_t)nn * KK;           // B row: K 0..511 = x part
  const __bf16* pbh = pb + DIN;                       //        K 512.. = h part
  const __bf16* pah = hbuf + (size_t)am * DH;
  const float gn = g_[nn], bn = bt_[nn], bias_n = bias[nn];
  unsigned lgen = 0;

  for (int t = 0; t < Ss; ++t) {
    v8f acc = {};
    const __bf16* pax = xb + ((size_t)am * Ss + t) * DIN;
#pragma unroll 4
    for (int kb = 0; kb < DIN; kb += 32)
      acc = wmma_bf16(load_a(pax + kb, lh), load_b(pb + kb, lh), acc);
#pragma unroll 4
    for (int kb = 0; kb < DH; kb += 32)
      acc = wmma_bf16(load_a(pah + kb, lh), load_b(pbh + kb, lh), acc);

#pragma unroll
    for (int r = 0; r < 8; ++r) acc[r] += bias_n;

    // per-row partial sums over this tile's 16 columns (shfl within 16-lane half)
    float* st = stats + (size_t)t * (Bb * 2);
#pragma unroll
    for (int r = 0; r < 8; ++r) {
      float v = acc[r];
      float s1 = v, s2 = v * v;
      for (int sh = 8; sh >= 1; sh >>= 1) {
        s1 += __shfl_xor(s1, sh, 32);
        s2 += __shfl_xor(s2, sh, 32);
      }
      if (tcol == 0) {
        int m = m0 + lh * 8 + r;
        atomicAdd(&st[m * 2 + 0], s1);
        atomicAdd(&st[m * 2 + 1], s2);
      }
    }
    grid_sync(&syncp[0], &syncp[1], lgen);            // stats complete, h reads done

#pragma unroll
    for (int r = 0; r < 8; ++r) {
      int m = m0 + lh * 8 + r;
      float s1 = st[m * 2 + 0], s2 = st[m * 2 + 1];
      float mu  = s1 * (1.0f / DH);
      float var = s2 * (1.0f / DH) - mu * mu;
      float y = (acc[r] - mu) * rsqrtf(var + 1e-5f) * gn + bn;
      y = fmaxf(y, 0.0f);
      hbuf[(size_t)m * DH + nn] = (__bf16)y;
      hsb[((size_t)m * Ss + t) * DH + nn] = (__bf16)y;
    }
    grid_sync(&syncp[0], &syncp[1], lgen);            // h fully published
  }
}

// ---------------------------------------------------------------------------
// Output GEMM: logits[8192][32000] = hs_bf16 @ WoutT_bf16 + bout (f32 acc).
// Wave = 2 M-tiles x 4 N-tiles (32 rows x 64 cols, 8 WMMA per k-step).
// Block = 8 waves = 64 rows x 256 cols; grid (125, 128).
// A/B operands are double-buffered: kb+32 is prefetched unconditionally (the
// final overread stays inside the workspace) so loads overlap the 8 WMMAs.
// WoutT (65 MB bf16) stays L2-resident; stores are non-temporal.
// ---------------------------------------------------------------------------
__global__ __launch_bounds__(256) void out_gemm(const __bf16* __restrict__ hsb,
                                                const __bf16* __restrict__ wt,   // [VV][DH]
                                                const float* __restrict__ bout,
                                                float* __restrict__ out) {
  const int lane = threadIdx.x & 31;
  const int wave = threadIdx.x >> 5;
  const int lh   = lane >> 4;
  const int tcol = lane & 15;
  const int r0 = blockIdx.y * 64 + (wave >> 2) * 32;          // 32 rows per wave
  const int c0 = blockIdx.x * 256 + (wave & 3) * 64;          // 64 cols per wave

  const __bf16* pa0 = hsb + (size_t)(r0 + tcol) * DH;         // M-tile 0 row
  const __bf16* pa1 = pa0 + (size_t)16 * DH;                  // M-tile 1 row
  const __bf16* pbn[4];
#pragma unroll
  for (int j = 0; j < 4; ++j) pbn[j] = wt + (size_t)(c0 + j * 16 + tcol) * DH;

  v8f acc[2][4] = {};

  // pipeline stage 0
  v16bf a0 = load_a(pa0, lh);
  v16bf a1 = load_a(pa1, lh);
  v16bf bm[4];
#pragma unroll
  for (int j = 0; j < 4; ++j) bm[j] = load_b(pbn[j], lh);

  for (int kb = 0; kb < DH; kb += 32) {
    // prefetch next k-step (last iteration overreads 64 B inside workspace)
    const int nk = kb + 32;
    v16bf na0 = load_a(pa0 + nk, lh);
    v16bf na1 = load_a(pa1 + nk, lh);
    v16bf nb[4];
#pragma unroll
    for (int j = 0; j < 4; ++j) nb[j] = load_b(pbn[j] + nk, lh);

#pragma unroll
    for (int j = 0; j < 4; ++j) {
      acc[0][j] = wmma_bf16(a0, bm[j], acc[0][j]);
      acc[1][j] = wmma_bf16(a1, bm[j], acc[1][j]);
    }
    a0 = na0; a1 = na1;
#pragma unroll
    for (int j = 0; j < 4; ++j) bm[j] = nb[j];
  }

#pragma unroll
  for (int j = 0; j < 4; ++j) {
    int n = c0 + j * 16 + tcol;
    float bv = bout[n];
#pragma unroll
    for (int mi = 0; mi < 2; ++mi) {
#pragma unroll
      for (int r = 0; r < 8; ++r) {
        int row = r0 + mi * 16 + lh * 8 + r;
        __builtin_nontemporal_store(acc[mi][j][r] + bv, &out[(size_t)row * VV + n]);
      }
    }
  }
}

// ---------------------------------------------------------------------------
// Per-row online logsumexp (streaming max + sum of exp), one block per row
// ---------------------------------------------------------------------------
__global__ __launch_bounds__(256) void row_lse(const float* __restrict__ logits,
                                               float* __restrict__ lse) {
  __shared__ float sm[256], ss[256];
  const float* row = logits + (size_t)blockIdx.x * VV;
  float m = -3.4e38f, s = 0.0f;
  for (int i = threadIdx.x; i < VV; i += 256) {
    float v = __builtin_nontemporal_load(row + i);
    if (v > m) { s = s * __expf(m - v) + 1.0f; m = v; }
    else       { s += __expf(v - m); }
  }
  sm[threadIdx.x] = m; ss[threadIdx.x] = s;
  __syncthreads();
  for (int w = 128; w > 0; w >>= 1) {
    if (threadIdx.x < w) {
      float m2 = sm[threadIdx.x + w], s2 = ss[threadIdx.x + w];
      float M = fmaxf(sm[threadIdx.x], m2);
      ss[threadIdx.x] = ss[threadIdx.x] * __expf(sm[threadIdx.x] - M) + s2 * __expf(m2 - M);
      sm[threadIdx.x] = M;
    }
    __syncthreads();
  }
  if (threadIdx.x == 0) lse[blockIdx.x] = sm[0] + __logf(ss[0]);
}

// ---------------------------------------------------------------------------
// out = logits - lse[row]    (vectorized; 32000 % 4 == 0 so v4f stays in-row)
// ---------------------------------------------------------------------------
__global__ __launch_bounds__(256) void finalize(float* __restrict__ out,
                                                const float* __restrict__ lse) {
  size_t i = ((size_t)blockIdx.x * 256 + threadIdx.x) * 4;
  float l = lse[(int)(i / VV)];
  v4f v = *(const v4f*)(out + i);
  v -= l;
  __builtin_nontemporal_store(v, (v4f*)(out + i));
}

// ---------------------------------------------------------------------------
// Launch. Workspace (~94 MB): WoutT | WT | x_bf16 | hs_bf16 | h | stats | lse | sync
// ---------------------------------------------------------------------------
extern "C" void kernel_launch(void* const* d_in, const int* in_sizes, int n_in,
                              void* d_out, int out_size, void* d_ws, size_t ws_size,
                              hipStream_t stream) {
  (void)in_sizes; (void)n_in; (void)out_size; (void)ws_size;
  const int*   tokens = (const int*)d_in[0];
  const float* emb    = (const float*)d_in[1];
  const float* W      = (const float*)d_in[2];
  const float* bvec   = (const float*)d_in[3];
  const float* gamma_ = (const float*)d_in[4];
  const float* beta_  = (const float*)d_in[5];
  const float* h0     = (const float*)d_in[6];
  const float* Wout   = (const float*)d_in[7];
  const float* bout   = (const float*)d_in[8];
  float* out = (float*)d_out;
  char*  ws  = (char*)d_ws;

  constexpr size_t oWoutT = 0;                                      // 65,536,000 B
  constexpr size_t oWT    = oWoutT + (size_t)VV * DH * 2;
  constexpr size_t oX     = oWT    + (size_t)DH * KK * 2;           // 3,145,728 B
  constexpr size_t oHS    = oX     + (size_t)ROWS * DIN * 2;        // 8,388,608 B
  constexpr size_t oH     = oHS    + (size_t)ROWS * DH * 2;         // 16,777,216 B
  constexpr size_t oST    = oH     + (size_t)Bb * DH * 2;           // 65,536 B
  constexpr size_t oLSE   = oST    + (size_t)Ss * Bb * 2 * 4;       // 65,536 B
  constexpr size_t oSYNC  = oLSE   + (size_t)ROWS * 4;              // 32,768 B

  __bf16*   woutT = (__bf16*)(ws + oWoutT);
  __bf16*   wT    = (__bf16*)(ws + oWT);
  __bf16*   xb    = (__bf16*)(ws + oX);
  __bf16*   hsb   = (__bf16*)(ws + oHS);
  __bf16*   hbuf  = (__bf16*)(ws + oH);
  float*    stats = (float*)(ws + oST);
  float*    lse   = (float*)(ws + oLSE);
  unsigned* syncp = (unsigned*)(ws + oSYNC);

  init_kernel <<<128, 256, 0, stream>>>(h0, hbuf, stats, syncp);
  embed_kernel<<<(ROWS * DIN) / 256, 256, 0, stream>>>(tokens, emb, xb);
  tconv<<<dim3(DH / 32, KK / 32), dim3(32, 8), 0, stream>>>(W, wT, KK, DH);
  tconv<<<dim3(VV / 32, DH / 32), dim3(32, 8), 0, stream>>>(Wout, woutT, DH, VV);
  rnn_scan<<<NWG, 256, 0, stream>>>(xb, wT, bvec, gamma_, beta_, hbuf, hsb, stats, syncp);
  out_gemm<<<dim3(VV / 256, ROWS / 64), 256, 0, stream>>>(hsb, woutT, bout, out);
  row_lse <<<ROWS, 256, 0, stream>>>(out, lse);
  finalize<<<(size_t)ROWS * VV / 1024, 256, 0, stream>>>(out, lse);
}
